// WordAttention_59167469469978
// MI455X (gfx1250) — matrix operand order repaired and checked by reading, not compiled, not run on hardware
//
#include <hip/hip_runtime.h>

// Problem constants (match reference)
#define B_  32
#define LP_ 2048
#define LQ_ 128
#define D_  768
#define H_  768

typedef __attribute__((ext_vector_type(16))) __bf16 v16bf;
typedef __attribute__((ext_vector_type(8)))  float  v8f;

// 16-bit A-matrix fragment (16x32): element i of v16bf for a lane with
// half = lane>>4 maps to K offset within the 32-wide K chunk:
//   K = half*16 + i   (contiguous, matches B fragment loads)
__device__ __forceinline__ int a_koff(int i, int half) {
  return (half << 4) + i;
}

// Load an A fragment from row-major bf16 LDS: two aligned b128 LDS loads.
__device__ __forceinline__ v16bf load_afrag_lds(const __bf16* base, int half) {
  union { v16bf v; uint4 q[2]; } r;
  const __bf16* p = base + (half << 4);
  r.q[0] = *(const uint4*)(p);
  r.q[1] = *(const uint4*)(p + 8);
  return r.v;
}

__device__ __forceinline__ v8f wmma_bf16(v16bf a, v16bf b, v8f c) {
  return __builtin_amdgcn_wmma_f32_16x16x32_bf16(false, a, false, b,
                                                 (short)0, c, false, false);
}

// ---------------------------------------------------------------------------
// Kernel 0a: elementwise fp32 -> bf16 (for W_w, row-major [H,D])
// ---------------------------------------------------------------------------
__global__ void cvt_bf16_kernel(const float* __restrict__ in,
                                __bf16* __restrict__ out, int n) {
  int i = blockIdx.x * blockDim.x + threadIdx.x;
  if (i < n) out[i] = (__bf16)in[i];
}

// ---------------------------------------------------------------------------
// Kernel 0b: question [B,LQ,D] fp32 -> Qt [B,D,LQ] bf16 (transposed)
// ---------------------------------------------------------------------------
__global__ void cvt_qt_kernel(const float* __restrict__ q,
                              __bf16* __restrict__ qt, int n) {
  int i = blockIdx.x * blockDim.x + threadIdx.x;  // i = ((b*D + d)*LQ + qq)
  if (i >= n) return;
  int qq = i % LQ_;
  int bd = i / LQ_;
  int d = bd % D_;
  int b = bd / D_;
  qt[i] = (__bf16)q[((size_t)b * LQ_ + qq) * D_ + d];
}

// ---------------------------------------------------------------------------
// Kernel 1: Wq = relu(question @ W^T + b)  -> bf16 [B*LQ, H]
// One wave per 16x16 output tile, K-loop over D in 32-wide chunks.
// ---------------------------------------------------------------------------
__global__ __launch_bounds__(256)
void proj_q_kernel(const float* __restrict__ question,
                   const __bf16* __restrict__ Wbf,
                   const float* __restrict__ Wb,
                   __bf16* __restrict__ Wq) {
  const int wave = (blockIdx.x * blockDim.x + threadIdx.x) >> 5;
  const int lane = threadIdx.x & 31;
  const int l15 = lane & 15, half = lane >> 4;
  const int NT = H_ / 16;                 // 48 n-tiles
  const int mtile = wave / NT;            // 0..255 (B*LQ/16)
  const int ntile = wave % NT;
  const int row0 = mtile * 16;
  const int h0 = ntile * 16;

  v8f c = {};
#pragma unroll 1
  for (int kc = 0; kc < D_ / 32; ++kc) {
    const int k0 = kc * 32;
    v16bf a;
    const float* qrow = question + (size_t)(row0 + l15) * D_ + k0;
#pragma unroll
    for (int i = 0; i < 16; ++i) a[i] = (__bf16)qrow[a_koff(i, half)];
    v16bf b = *(const v16bf*)(Wbf + (size_t)(h0 + l15) * D_ + k0 + half * 16);
    c = wmma_bf16(a, b, c);
  }
  const float bias = Wb[h0 + l15];
#pragma unroll
  for (int r = 0; r < 8; ++r) {
    float v = fmaxf(c[r] + bias, 0.0f);
    const int m = r + (half << 3);
    Wq[(size_t)(row0 + m) * H_ + h0 + l15] = (__bf16)v;
  }
}

// ---------------------------------------------------------------------------
// Kernel 2: fused  relu(P@W^T+b) -> scores -> masked softmax -> alpha@Qt
// One wave per 16-row passage tile; passage staged once in LDS as bf16.
// 2 waves / block (58 KB static LDS; ~5 blocks co-resident per 320KB WGP).
// ---------------------------------------------------------------------------
#define WPB 2
__global__ __launch_bounds__(64)
void attn_kernel(const float* __restrict__ passage,
                 const unsigned char* __restrict__ qmask,
                 const __bf16* __restrict__ Wbf,
                 const float* __restrict__ Wb,
                 const __bf16* __restrict__ Wq,
                 const __bf16* __restrict__ Qt,
                 float* __restrict__ out) {
  __shared__ alignas(16) __bf16 sP[WPB][16 * 768];   // passage tile, bf16
  __shared__ alignas(16) __bf16 sWp[WPB][16 * 32];   // projected chunk
  __shared__ alignas(16) __bf16 sAl[WPB][16 * 128];  // alpha tile

  const int wv = threadIdx.x >> 5;
  const int lane = threadIdx.x & 31;
  const int l15 = lane & 15, half = lane >> 4;

  const int tile = blockIdx.x * WPB + wv;   // 0..4095
  const int b = tile >> 7;                  // LP/16 = 128 tiles per batch
  const int mt = tile & 127;
  const int prow0 = mt * 16;

  const float* P = passage + ((size_t)b * LP_ + prow0) * D_;
  const __bf16* WQb = Wq + (size_t)b * LQ_ * H_;
  const __bf16* Qtb = Qt + (size_t)b * D_ * LQ_;

  // ---- phase 0: stage passage tile -> LDS bf16 ----
  // 8 floats / lane / iter: 2x coalesced b128 global loads -> 1x b128 DS store
#pragma unroll 1
  for (int r = 0; r < 16; ++r) {
    const float4* prow = (const float4*)(P + (size_t)r * D_);
#pragma unroll
    for (int it = 0; it < 3; ++it) {          // 3 * 32 lanes * 8 floats = 768
      const int c8 = it * 32 + lane;
      float4 v0 = prow[c8 * 2];
      float4 v1 = prow[c8 * 2 + 1];
      union { uint4 u; __bf16 e[8]; } pk;
      pk.e[0] = (__bf16)v0.x; pk.e[1] = (__bf16)v0.y;
      pk.e[2] = (__bf16)v0.z; pk.e[3] = (__bf16)v0.w;
      pk.e[4] = (__bf16)v1.x; pk.e[5] = (__bf16)v1.y;
      pk.e[6] = (__bf16)v1.z; pk.e[7] = (__bf16)v1.w;
      *(uint4*)&sP[wv][r * 768 + c8 * 8] = pk.u;
    }
  }
  __syncthreads();

  bool mk[8];
#pragma unroll
  for (int qt = 0; qt < 8; ++qt)
    mk[qt] = qmask[(size_t)b * LQ_ + qt * 16 + l15] != 0;

  const v8f vz = {};
  v8f sc[8];
#pragma unroll
  for (int qt = 0; qt < 8; ++qt) sc[qt] = vz;

  // ---- scores[16x128] = relu(P_tile @ W^T + b) @ Wq^T, chunked over H ----
#pragma unroll 1
  for (int hc = 0; hc < H_ / 32; ++hc) {
    const int hb = hc * 32;
    v8f c0 = vz, c1 = vz;
#pragma unroll 1
    for (int kc = 0; kc < D_ / 32; ++kc) {
      const int k0 = kc * 32;
      v16bf a = load_afrag_lds(&sP[wv][l15 * 768 + k0], half);
      v16bf b0 = *(const v16bf*)(Wbf + (size_t)(hb + l15) * D_ + k0 + half * 16);
      v16bf b1 = *(const v16bf*)(Wbf + (size_t)(hb + 16 + l15) * D_ + k0 + half * 16);
      c0 = wmma_bf16(a, b0, c0);
      c1 = wmma_bf16(a, b1, c1);
    }
    const float bias0 = Wb[hb + l15];
    const float bias1 = Wb[hb + 16 + l15];
#pragma unroll
    for (int r = 0; r < 8; ++r) {
      const int m = r + (half << 3);
      sWp[wv][m * 32 + l15]      = (__bf16)fmaxf(c0[r] + bias0, 0.0f);
      sWp[wv][m * 32 + 16 + l15] = (__bf16)fmaxf(c1[r] + bias1, 0.0f);
    }
    __syncthreads();                      // C-layout -> row-major in LDS
    v16bf aw = load_afrag_lds(&sWp[wv][l15 * 32], half);
    __syncthreads();                      // reads done before next overwrite
#pragma unroll
    for (int qt = 0; qt < 8; ++qt) {
      v16bf bq = *(const v16bf*)(WQb + (size_t)(qt * 16 + l15) * H_ + hb + half * 16);
      sc[qt] = wmma_bf16(aw, bq, sc[qt]);
    }
  }

  // ---- mask + softmax over q (rows span lanes 0-15 / 16-31 per half) ----
  const float NEG = -__builtin_huge_valf();
#pragma unroll
  for (int qt = 0; qt < 8; ++qt)
    if (mk[qt]) {
#pragma unroll
      for (int r = 0; r < 8; ++r) sc[qt][r] = NEG;
    }
#pragma unroll
  for (int r = 0; r < 8; ++r) {
    float mx = NEG;
#pragma unroll
    for (int qt = 0; qt < 8; ++qt) mx = fmaxf(mx, sc[qt][r]);
#pragma unroll
    for (int s = 1; s < 16; s <<= 1) mx = fmaxf(mx, __shfl_xor(mx, s, 32));
    float sum = 0.0f;
#pragma unroll
    for (int qt = 0; qt < 8; ++qt) {
      const float e = __expf(sc[qt][r] - mx);
      sc[qt][r] = e;
      sum += e;
    }
#pragma unroll
    for (int s = 1; s < 16; s <<= 1) sum += __shfl_xor(sum, s, 32);
    const float inv = 1.0f / sum;
    const int m = r + (half << 3);
#pragma unroll
    for (int qt = 0; qt < 8; ++qt)
      sAl[wv][m * 128 + qt * 16 + l15] = (__bf16)(sc[qt][r] * inv);
  }
  __syncthreads();

  v16bf aal[4];
#pragma unroll
  for (int kc = 0; kc < 4; ++kc)
    aal[kc] = load_afrag_lds(&sAl[wv][l15 * 128 + kc * 32], half);

  // ---- out[16x768] = alpha[16x128] @ question[128x768]  (Qt is [D,LQ]) ----
  float* O = out + ((size_t)b * LP_ + prow0) * D_;
#pragma unroll 1
  for (int nt = 0; nt < D_ / 16; ++nt) {
    v8f o = vz;
#pragma unroll
    for (int kc = 0; kc < 4; ++kc) {
      v16bf bq = *(const v16bf*)(Qtb + (size_t)(nt * 16 + l15) * LQ_ + kc * 32 + half * 16);
      o = wmma_bf16(aal[kc], bq, o);
    }
#pragma unroll
    for (int r = 0; r < 8; ++r)
      O[(size_t)(r + (half << 3)) * D_ + nt * 16 + l15] = o[r];
  }
}

// ---------------------------------------------------------------------------
// Launch: inputs = passage, p_mask, question, q_mask, W_w, W_b
// Workspace: Wbf[H*D] | Qt[B*D*LQ] | Wq[B*LQ*H]  all bf16  (~13.8 MB)
// ---------------------------------------------------------------------------
extern "C" void kernel_launch(void* const* d_in, const int* in_sizes, int n_in,
                              void* d_out, int out_size, void* d_ws, size_t ws_size,
                              hipStream_t stream) {
  const float* passage  = (const float*)d_in[0];
  const float* question = (const float*)d_in[2];
  const unsigned char* q_mask = (const unsigned char*)d_in[3];
  const float* W_w = (const float*)d_in[4];
  const float* W_b = (const float*)d_in[5];
  float* out = (float*)d_out;

  constexpr int   Wn = H_ * D_;             // 589824
  constexpr int   Qn = B_ * D_ * LQ_;       // 3145728
  constexpr size_t WbfOff = 0;
  constexpr size_t QtOff  = WbfOff + (size_t)Wn * 2;   // 1,179,648
  constexpr size_t WqOff  = QtOff + (size_t)Qn * 2;    // 7,471,104
  __bf16* Wbf = (__bf16*)((char*)d_ws + WbfOff);
  __bf16* Qt  = (__bf16*)((char*)d_ws + QtOff);
  __bf16* Wq  = (__bf16*)((char*)d_ws + WqOff);

  cvt_bf16_kernel<<<(Wn + 255) / 256, 256, 0, stream>>>(W_w, Wbf, Wn);
  cvt_qt_kernel<<<(Qn + 255) / 256, 256, 0, stream>>>(question, Qt, Qn);

  // (B*LQ/16) * (H/16) = 256*48 = 12288 waves, 8 waves/block
  proj_q_kernel<<<12288 / 8, 256, 0, stream>>>(question, Wbf, W_b, Wq);

  // B * LP/16 = 4096 tiles, 2 waves/block
  attn_kernel<<<4096 / WPB, 64, 0, stream>>>(passage, q_mask, Wbf, W_b,
                                             Wq, Qt, out);
}
